// FairRep_66194035966015
// MI455X (gfx1250) — matrix operand order repaired and checked by reading, not compiled, not run on hardware
//
#include <hip/hip_runtime.h>
#include <hip/hip_bf16.h>
#include <math.h>

// FairRep forward on MI455X (gfx1250, wave32, WMMA).
// GEMMs use v_wmma_f32_16x16x32_f16; batch-norm stats via LDS + global f32 atomics.
// Workspace usage ~775 MB (t1 f32, staged f16 activations, 4 encoder outputs).

#define NS 262144

typedef __attribute__((ext_vector_type(16))) _Float16 v16h;
typedef __attribute__((ext_vector_type(8)))  float    v8f;

union FragH { v16h h; unsigned u[8]; };
union PackH { unsigned u; _Float16 h[2]; };

// A-matrix 16x32 f16 K-pattern: VGPR v, half hh (lane>=16)
__device__ __forceinline__ int kpatA(int v, int hh) {
    return ((v < 4) ? (2 * v) : (8 + 2 * v)) + 8 * hh;
}
// B-matrix 32x16 f16 K-pattern (column-major per lane)
__device__ __forceinline__ int kpatB(int v, int hh) {
    return ((v < 4) ? (2 * v) : (8 + 2 * v)) + 8 * hh;
}

__device__ __forceinline__ float sigmoidf_(float v) { return 1.f / (1.f + expf(-v)); }
__device__ __forceinline__ float softplusf_(float v) {
    return fmaxf(v, 0.f) + log1pf(expf(-fabsf(v)));
}
// t*log(sigmoid(u)) + (1-t)*log(1-sigmoid(u))
__device__ __forceinline__ float bce_term(float u, float t) {
    return -(t * softplusf_(-u) + (1.f - t) * softplusf_(u));
}

// ---------------------------------------------------------------------------
// zero scratch scalars
__global__ void zero_f32(float* p, int n) {
    for (int i = blockIdx.x * blockDim.x + threadIdx.x; i < n; i += gridDim.x * blockDim.x)
        p[i] = 0.f;
}

// ---------------------------------------------------------------------------
// Fold weights: transpose to col-major f16, collapse cla (10->2) and dec (20->128).
__global__ void prep_weights(const float* __restrict__ W1, const float* __restrict__ W2,
                             const float* __restrict__ Wn, const float* __restrict__ b2,
                             const float* __restrict__ Wc1, const float* __restrict__ bc1,
                             const float* __restrict__ Wc2, const float* __restrict__ bc2,
                             const float* __restrict__ Wd1, const float* __restrict__ bd1,
                             const float* __restrict__ Wd2, const float* __restrict__ bd2,
                             _Float16* __restrict__ W1t, _Float16* __restrict__ W2t,
                             _Float16* __restrict__ Wnt, float* __restrict__ b2p,
                             float* __restrict__ WcE, float* __restrict__ bcE,
                             float* __restrict__ WdE, float* __restrict__ bdE) {
    int tid = threadIdx.x;
    for (int idx = tid; idx < 128 * 128; idx += 256) {
        int n = idx >> 7, k = idx & 127;
        W1t[n * 128 + k] = (_Float16)W1[k * 128 + n];
    }
    for (int idx = tid; idx < 48 * 128; idx += 256) {
        int n = idx >> 7, k = idx & 127;
        W2t[n * 128 + k] = (_Float16)((n < 40) ? W2[k * 40 + n] : 0.f);
    }
    for (int idx = tid; idx < 128 * 160; idx += 256) {
        int n = idx / 160, k = idx - n * 160;
        Wnt[n * 160 + k] = (_Float16)((k < 129) ? Wn[k * 128 + n] : 0.f);
    }
    if (tid < 48) b2p[tid] = (tid < 40) ? b2[tid] : 0.f;
    if (tid < 20) {
        for (int c = 0; c < 2; ++c) {
            float s = 0.f;
            for (int j = 0; j < 32; ++j) s += Wc1[tid * 32 + j] * Wc2[j * 2 + c];
            WcE[tid * 2 + c] = s;
        }
    }
    if (tid < 2) {
        float s = bc2[tid];
        for (int j = 0; j < 32; ++j) s += bc1[j] * Wc2[j * 2 + tid];
        bcE[tid] = s;
    }
    for (int idx = tid; idx < 20 * 128; idx += 256) {
        int i2 = idx >> 7, j = idx & 127;
        float s = 0.f;
        for (int k = 0; k < 16; ++k) s += Wd1[i2 * 16 + k] * Wd2[k * 128 + j];
        WdE[idx] = s;
    }
    if (tid < 128) {
        float s = bd2[tid];
        for (int k = 0; k < 16; ++k) s += bd1[k] * Wd2[k * 128 + tid];
        bdE[tid] = s;
    }
}

// ---------------------------------------------------------------------------
// Build [x|a] and [x|1-a] f16, lda=128
__global__ void build_X12(const float* __restrict__ x, const float* __restrict__ a,
                          _Float16* __restrict__ X1h, _Float16* __restrict__ X2h) {
    for (long long g = (long long)blockIdx.x * 256 + threadIdx.x; g < (long long)NS * 128;
         g += (long long)gridDim.x * 256) {
        int i = (int)(g >> 7), j = (int)(g & 127);
        float v1, v2;
        if (j < 126) { float xv = x[(size_t)i * 126 + j]; v1 = xv; v2 = xv; }
        else { float av = a[(size_t)i * 2 + (j - 126)]; v1 = av; v2 = 1.f - av; }
        X1h[g] = (_Float16)v1; X2h[g] = (_Float16)v2;
    }
}

// Build [x|a|ypred|0-pad] and [x|1-a|ypred|0-pad] f16, lda=160 (K padded for WMMA)
__global__ void build_Xn(const float* __restrict__ x, const float* __restrict__ a,
                         const float* __restrict__ yp,
                         _Float16* __restrict__ Xn1h, _Float16* __restrict__ Xn2h) {
    for (long long g = (long long)blockIdx.x * 256 + threadIdx.x; g < (long long)NS * 160;
         g += (long long)gridDim.x * 256) {
        int i = (int)(g / 160), j = (int)(g - (long long)i * 160);
        float v1, v2;
        if (j < 126) { float xv = x[(size_t)i * 126 + j]; v1 = xv; v2 = xv; }
        else if (j < 128) { float av = a[(size_t)i * 2 + (j - 126)]; v1 = av; v2 = 1.f - av; }
        else if (j == 128) { float pv = yp[i]; v1 = pv; v2 = pv; }
        else { v1 = 0.f; v2 = 0.f; }
        Xn1h[g] = (_Float16)v1; Xn2h[g] = (_Float16)v2;
    }
}

// ---------------------------------------------------------------------------
// Encoder stage 1: t1 = A(N,128) @ W1 + b1, f32 out, + per-column sum/sumsq atomics.
// Each wave: 16x16 tile. Col-tile fixed per block (8 waves share) -> cheap stat reduce.
__global__ void __launch_bounds__(256)
gemm128_stats(const _Float16* __restrict__ A, const _Float16* __restrict__ Bt,
              const float* __restrict__ bias, float* __restrict__ Out,
              float* __restrict__ csum, float* __restrict__ csq) {
    const int tid = threadIdx.x, wid = tid >> 5, lane = tid & 31;
    const int l = lane & 15, hh = lane >> 4;
    const int j = blockIdx.x & 7;
    const int col = j * 16 + l;

    FragH bf[4];
    {
        const _Float16* bp = Bt + (size_t)col * 128;
        #pragma unroll
        for (int kc = 0; kc < 4; ++kc)
            #pragma unroll
            for (int v = 0; v < 8; ++v)
                bf[kc].u[v] = *(const unsigned*)(bp + kc * 32 + kpatB(v, hh));
    }
    const float bcol = bias[col];
    float lsum = 0.f, lsq = 0.f;
    const int wglobal = (blockIdx.x >> 3) * 8 + wid;
    const int wstride = (gridDim.x >> 3) * 8;
    float* const obase = Out + col + (size_t)8 * 128 * hh;
    for (int rt = wglobal; rt < (NS / 16); rt += wstride) {
        FragH af[4];
        const _Float16* ap = A + (size_t)(rt * 16 + l) * 128;
        #pragma unroll
        for (int kc = 0; kc < 4; ++kc)
            #pragma unroll
            for (int v = 0; v < 8; ++v)
                af[kc].u[v] = *(const unsigned*)(ap + kc * 32 + kpatA(v, hh));
        v8f c;
        #pragma unroll
        for (int r = 0; r < 8; ++r) c[r] = bcol;
        #pragma unroll
        for (int kc = 0; kc < 4; ++kc)
            c = __builtin_amdgcn_wmma_f32_16x16x32_f16(false, af[kc].h, false, bf[kc].h,
                                                       (short)0, c, false, false);
        float* po = obase + (size_t)rt * 16 * 128;   // constant-offset stores below
        #pragma unroll
        for (int r = 0; r < 8; ++r) {
            float v = c[r];
            po[r * 128] = v;
            lsum += v; lsq += v * v;
        }
    }
    __shared__ float ssum[16], ssq[16];
    if (tid < 16) { ssum[tid] = 0.f; ssq[tid] = 0.f; }
    __syncthreads();
    atomicAdd(&ssum[l], lsum);
    atomicAdd(&ssq[l], lsq);
    __syncthreads();
    if (tid < 16) {
        atomicAdd(&csum[j * 16 + tid], ssum[tid]);
        atomicAdd(&csq[j * 16 + tid], ssq[tid]);
    }
}

// BN coefficients: a = g*rsqrt(var+eps), c = be - mu*a  (biased var, matches torch BN)
__global__ void bn_finalize(const float* __restrict__ csum, const float* __restrict__ csq,
                            const float* __restrict__ g, const float* __restrict__ be,
                            float* __restrict__ aOut, float* __restrict__ cOut) {
    int i = threadIdx.x;
    float mu = csum[i] * (1.f / (float)NS);
    float var = csq[i] * (1.f / (float)NS) - mu * mu;
    float av = g[i] * rsqrtf(var + 1e-5f);
    aOut[i] = av;
    cOut[i] = be[i] - mu * av;
}

// Column stats of relu(a1*t1+c1) (needed for the second BatchNorm)
__global__ void __launch_bounds__(256)
relu_bn_stats(const float* __restrict__ T1, const float* __restrict__ aC,
              const float* __restrict__ cC, float* __restrict__ csum2,
              float* __restrict__ csq2) {
    int tid = threadIdx.x;
    int gt = blockIdx.x * 256 + tid;
    int col = gt & 127;
    float av = aC[col], cv = cC[col];
    float ls = 0.f, lq = 0.f;
    int stride = (gridDim.x * 256) >> 7;
    const float* p = T1 + col + (size_t)(gt >> 7) * 128;
    const float* pend = T1 + col + (size_t)NS * 128;
    for (; p < pend; p += (size_t)stride * 128) {
        float v = fmaxf(av * (*p) + cv, 0.f);
        ls += v; lq += v * v;
    }
    __shared__ float ss[128], sq[128];
    if (tid < 128) { ss[tid] = 0.f; sq[tid] = 0.f; }
    __syncthreads();
    atomicAdd(&ss[tid & 127], ls);
    atomicAdd(&sq[tid & 127], lq);
    __syncthreads();
    if (tid < 128) {
        atomicAdd(&csum2[tid], ss[tid]);
        atomicAdd(&csq2[tid], sq[tid]);
    }
}

// Encoder head: E = relu(bn2(relu(bn1(t1)))) @ W2 + b2  (48 padded cols, 40 valid)
__global__ void __launch_bounds__(256)
gemm_bn_out(const float* __restrict__ T1, const float* __restrict__ A1,
            const float* __restrict__ C1, const float* __restrict__ A2,
            const float* __restrict__ C2, const _Float16* __restrict__ W2t,
            const float* __restrict__ b2p, float* __restrict__ E) {
    __shared__ float sa1[128], sc1[128], sa2[128], sc2[128];
    for (int i = threadIdx.x; i < 128; i += 256) {
        sa1[i] = A1[i]; sc1[i] = C1[i]; sa2[i] = A2[i]; sc2[i] = C2[i];
    }
    __syncthreads();
    const int tid = threadIdx.x, wid = tid >> 5, lane = tid & 31;
    const int l = lane & 15, hh = lane >> 4;
    const int nTiles = (NS / 16) * 3;
    for (int t = blockIdx.x * 8 + wid; t < nTiles; t += gridDim.x * 8) {
        int rt = t / 3, ct = t - rt * 3;
        int col = ct * 16 + l;
        FragH bf[4], af[4];
        const _Float16* bp = W2t + (size_t)col * 128;
        const float* ap = T1 + (size_t)(rt * 16 + l) * 128;
        #pragma unroll
        for (int kc = 0; kc < 4; ++kc) {
            #pragma unroll
            for (int v = 0; v < 8; ++v) {
                bf[kc].u[v] = *(const unsigned*)(bp + kc * 32 + kpatB(v, hh));
                int k = kc * 32 + kpatA(v, hh);
                float2 x2 = *(const float2*)(ap + k);
                float u0 = fmaxf(fmaxf(x2.x * sa1[k] + sc1[k], 0.f) * sa2[k] + sc2[k], 0.f);
                float u1 = fmaxf(fmaxf(x2.y * sa1[k + 1] + sc1[k + 1], 0.f) * sa2[k + 1] + sc2[k + 1], 0.f);
                PackH p; p.h[0] = (_Float16)u0; p.h[1] = (_Float16)u1;
                af[kc].u[v] = p.u;
            }
        }
        v8f c;
        float bcol = b2p[col];
        #pragma unroll
        for (int r = 0; r < 8; ++r) c[r] = bcol;
        #pragma unroll
        for (int kc = 0; kc < 4; ++kc)
            c = __builtin_amdgcn_wmma_f32_16x16x32_f16(false, af[kc].h, false, bf[kc].h,
                                                       (short)0, c, false, false);
        float* po = E + (size_t)(rt * 16 + 8 * hh) * 48 + col;
        #pragma unroll
        for (int r = 0; r < 8; ++r)
            po[r * 48] = c[r];
    }
}

// new_net: H = relu(Xn(N,160) @ Wn + bnb), f16 out for the next encoder GEMM
__global__ void __launch_bounds__(256)
gemm160_relu_h(const _Float16* __restrict__ A, const _Float16* __restrict__ Bt,
               const float* __restrict__ bias, _Float16* __restrict__ Out) {
    const int tid = threadIdx.x, wid = tid >> 5, lane = tid & 31;
    const int l = lane & 15, hh = lane >> 4;
    const int j = blockIdx.x & 7;
    const int col = j * 16 + l;
    FragH bf[5];
    {
        const _Float16* bp = Bt + (size_t)col * 160;
        #pragma unroll
        for (int kc = 0; kc < 5; ++kc)
            #pragma unroll
            for (int v = 0; v < 8; ++v)
                bf[kc].u[v] = *(const unsigned*)(bp + kc * 32 + kpatB(v, hh));
    }
    const float bcol = bias[col];
    const int wglobal = (blockIdx.x >> 3) * 8 + wid;
    const int wstride = (gridDim.x >> 3) * 8;
    _Float16* const obase = Out + col + (size_t)8 * 128 * hh;
    for (int rt = wglobal; rt < (NS / 16); rt += wstride) {
        FragH af[5];
        const _Float16* ap = A + (size_t)(rt * 16 + l) * 160;
        #pragma unroll
        for (int kc = 0; kc < 5; ++kc)
            #pragma unroll
            for (int v = 0; v < 8; ++v)
                af[kc].u[v] = *(const unsigned*)(ap + kc * 32 + kpatA(v, hh));
        v8f c;
        #pragma unroll
        for (int r = 0; r < 8; ++r) c[r] = bcol;
        #pragma unroll
        for (int kc = 0; kc < 5; ++kc)
            c = __builtin_amdgcn_wmma_f32_16x16x32_f16(false, af[kc].h, false, bf[kc].h,
                                                       (short)0, c, false, false);
        _Float16* po = obase + (size_t)rt * 16 * 128;
        #pragma unroll
        for (int r = 0; r < 8; ++r)
            po[r * 128] = (_Float16)fmaxf(c[r], 0.f);
    }
}

// ---------------------------------------------------------------------------
// y_pred = argmax(cla(zx1)+cla(zx2)) with folded 10->2 classifier
__global__ void ypred_kernel(const float* __restrict__ E1, const float* __restrict__ E2,
                             const float* __restrict__ eps, const float* __restrict__ WcE,
                             const float* __restrict__ bcE, float* __restrict__ yp) {
    int i = blockIdx.x * 256 + threadIdx.x;
    if (i >= NS) return;
    const float* r1 = E1 + (size_t)i * 48;
    const float* r2 = E2 + (size_t)i * 48;
    const float* e0 = eps + (size_t)i * 10;
    const float* e2 = eps + (size_t)2 * NS * 10 + (size_t)i * 10;
    float lg0 = 2.f * bcE[0], lg1 = 2.f * bcE[1];
    #pragma unroll
    for (int d = 0; d < 10; ++d) {
        float z1 = e0[d] * expf(0.5f * r1[20 + d]) + r1[d];
        float z2 = e2[d] * expf(0.5f * r2[20 + d]) + r2[d];
        float zs = z1 + z2;
        lg0 += zs * WcE[d * 2 + 0];
        lg1 += zs * WcE[d * 2 + 1];
    }
    yp[i] = (lg1 > lg0) ? 1.f : 0.f;
}

// ---------------------------------------------------------------------------
// Per-sample losses: KLs, reparam, folded decoder + BCE, CE, twin sums, corr sums.
// accum layout: 0 Scl,1 Sat,2 Scl2,3 Satcl,4 Szx,5 Szs,6 Sxs1,7 Sxs2,
//               8 Srx,9 Sra,10 Ssr, 12..15 cnt[4], 16..55 s1[4][10], 56..95 s2[4][10]
__global__ void __launch_bounds__(256)
final_losses(const float* __restrict__ E3, const float* __restrict__ E4,
             const float* __restrict__ eps, const float* __restrict__ x,
             const float* __restrict__ a, const int* __restrict__ y,
             const float* __restrict__ WdE, const float* __restrict__ bdE,
             const float* __restrict__ WcE, const float* __restrict__ bcE,
             float* __restrict__ accum) {
    __shared__ float sWd[2560];
    __shared__ float sbd[128];
    __shared__ float sWc[20], sbc[2];
    __shared__ float sacc[96];
    int tid = threadIdx.x;
    for (int i = tid; i < 2560; i += 256) sWd[i] = WdE[i];
    if (tid < 128) sbd[tid] = bdE[tid];
    if (tid < 20) sWc[tid] = WcE[tid];
    if (tid < 2) sbc[tid] = bcE[tid];
    if (tid < 96) sacc[tid] = 0.f;
    __syncthreads();
    int i = blockIdx.x * 256 + tid;
    if (i < NS) {
        const float* r3 = E3 + (size_t)i * 48;
        const float* r4 = E4 + (size_t)i * 48;
        float mx1[10], ms1[10], lx1[10], ls1[10];
        float mx2[10], ms2[10], lx2[10], ls2[10];
        #pragma unroll
        for (int d = 0; d < 10; ++d) {
            mx1[d] = r3[d]; ms1[d] = r3[10 + d]; lx1[d] = r3[20 + d]; ls1[d] = r3[30 + d];
            mx2[d] = r4[d]; ms2[d] = r4[10 + d]; lx2[d] = r4[20 + d]; ls2[d] = r4[30 + d];
        }
        // symmetric KL per-sample contributions: sym = 0.025*T - 0.5, T = sum of both dirs
        float Tzx = 0.f, Tzs = 0.f, Txs1 = 0.f, Txs2 = 0.f;
        #pragma unroll
        for (int d = 0; d < 10; ++d) {
            float dl, dm;
            dl = lx1[d] - lx2[d]; dm = mx1[d] - mx2[d];
            Tzx += expf(dl) + expf(-dl) + dm * dm * (expf(-lx2[d]) + expf(-lx1[d]));
            dl = ls1[d] - ls2[d]; dm = ms1[d] - ms2[d];
            Tzs += expf(dl) + expf(-dl) + dm * dm * (expf(-ls2[d]) + expf(-ls1[d]));
            dl = lx1[d] - ls1[d]; dm = mx1[d] - ms1[d];
            Txs1 += expf(dl) + expf(-dl) + dm * dm * (expf(-ls1[d]) + expf(-lx1[d]));
            dl = lx2[d] - ls2[d]; dm = mx2[d] - ms2[d];
            Txs2 += expf(dl) + expf(-dl) + dm * dm * (expf(-ls2[d]) + expf(-lx2[d]));
        }
        float c_zx = 0.025f * Tzx - 0.5f;
        float c_zs = 0.025f * Tzs - 0.5f;
        float c_xs1 = 0.025f * Txs1 - 0.5f;
        float c_xs2 = 0.025f * Txs2 - 0.5f;
        // reparameterized latents (pass 2)
        const float* e4 = eps + (size_t)4 * NS * 10 + (size_t)i * 10;
        const float* e5 = eps + (size_t)5 * NS * 10 + (size_t)i * 10;
        const float* e6 = eps + (size_t)6 * NS * 10 + (size_t)i * 10;
        const float* e7 = eps + (size_t)7 * NS * 10 + (size_t)i * 10;
        float zc1[20], zc2[20], zc3[20], zc4[20], zx1[10];
        #pragma unroll
        for (int d = 0; d < 10; ++d) {
            float vzx1 = e4[d] * expf(0.5f * lx1[d]) + mx1[d];
            float vzs1 = e5[d] * expf(0.5f * ls1[d]) + ms1[d];
            float vzx2 = e6[d] * expf(0.5f * lx2[d]) + mx2[d];
            float vzs2 = e7[d] * expf(0.5f * ls2[d]) + ms2[d];
            zx1[d] = vzx1;
            zc1[d] = vzx1; zc1[10 + d] = vzs1;   // r1  = dec([zx1,zs1])
            zc2[d] = vzx2; zc2[10 + d] = vzs2;   // r2  = dec([zx2,zs2])
            zc3[d] = vzx1; zc3[10 + d] = vzs2;   // rs1 = dec([zx1,zs2])
            zc4[d] = vzx2; zc4[10 + d] = vzs1;   // rs2 = dec([zx2,zs1])
        }
        // decoder (folded 20x128) + stable BCE accumulation
        float a0 = a[(size_t)i * 2], a1v = a[(size_t)i * 2 + 1];
        float Srx = 0.f, Sra = 0.f, Ssr = 0.f;
        const float* xr = x + (size_t)i * 126;
        for (int j = 0; j < 128; ++j) {
            float u1 = sbd[j], u2 = u1, u3 = u1, u4 = u1;
            #pragma unroll
            for (int k = 0; k < 20; ++k) {
                float w = sWd[k * 128 + j];
                u1 += zc1[k] * w; u2 += zc2[k] * w;
                u3 += zc3[k] * w; u4 += zc4[k] * w;
            }
            if (j < 126) {
                float t = sigmoidf_(xr[j]);
                Srx += bce_term(u1, t) + bce_term(u2, t);
                Ssr += bce_term(u3, t) + bce_term(u4, t);
            } else {
                float av = (j == 126) ? a0 : a1v;
                float ta = sigmoidf_(av), tb = sigmoidf_(1.f - av);
                Sra += bce_term(u1, ta) + bce_term(u2, ta);
                Ssr += bce_term(u3, ta) + bce_term(u4, tb);
            }
        }
        // folded classifier CE (log-softmax over 2 classes)
        float l10 = sbc[0], l11 = sbc[1], l20 = sbc[0], l21 = sbc[1];
        #pragma unroll
        for (int d = 0; d < 10; ++d) {
            l10 += zc1[d] * sWc[d * 2 + 0]; l11 += zc1[d] * sWc[d * 2 + 1];
            l20 += zc2[d] * sWc[d * 2 + 0]; l21 += zc2[d] * sWc[d * 2 + 1];
        }
        int yi = y[i];
        float m1 = fmaxf(l10, l11);
        float ce1 = m1 + logf(expf(l10 - m1) + expf(l11 - m1)) - ((yi == 0) ? l10 : l11);
        float m2 = fmaxf(l20, l21);
        float ce2 = m2 + logf(expf(l20 - m2) + expf(l21 - m2)) - ((yi == 0) ? l20 : l21);
        float cl = 0.5f * (ce1 + ce2);
        int at = (a1v > a0) ? 1 : 0;
        float atf = (float)at;
        int gid = yi * 2 + at;
        atomicAdd(&sacc[0], cl);
        atomicAdd(&sacc[1], atf);
        atomicAdd(&sacc[2], cl * cl);
        atomicAdd(&sacc[3], atf * cl);
        atomicAdd(&sacc[4], c_zx);
        atomicAdd(&sacc[5], c_zs);
        atomicAdd(&sacc[6], c_xs1);
        atomicAdd(&sacc[7], c_xs2);
        atomicAdd(&sacc[8], Srx);
        atomicAdd(&sacc[9], Sra);
        atomicAdd(&sacc[10], Ssr);
        atomicAdd(&sacc[12 + gid], 1.f);
        #pragma unroll
        for (int d = 0; d < 10; ++d) {
            atomicAdd(&sacc[16 + gid * 10 + d], zx1[d]);
            atomicAdd(&sacc[56 + gid * 10 + d], zx1[d] * zx1[d]);
        }
    }
    __syncthreads();
    if (tid < 96) atomicAdd(&accum[tid], sacc[tid]);
}

// ---------------------------------------------------------------------------
__global__ void finalize_out(const float* __restrict__ accum, float* __restrict__ out) {
    if (threadIdx.x != 0 || blockIdx.x != 0) return;
    const float Nf = (float)NS;
    float Scl = accum[0], Sat = accum[1], Scl2 = accum[2], Satcl = accum[3];
    out[0] = Scl;                                        // sum(cl)
    float xm = Sat / Nf, ym = Scl / Nf;
    float xv = (Sat - Nf * xm * xm) / (Nf - 1.f);        // Sxx == Sat for 0/1 tags
    float yv = (Scl2 - Nf * ym * ym) / (Nf - 1.f);
    float num = Satcl - Nf * xm * ym;
    float c = fabsf(num / sqrtf(xv * yv));
    out[1] = (xm == 0.f) ? 0.f : c;                      // F1 = 1.0
    float twin = 0.f;
    for (int yb = 0; yb < 2; ++yb) {
        int g0 = yb * 2, g1 = yb * 2 + 1;
        float c0 = accum[12 + g0], c1v = accum[12 + g1];
        float dsum = 0.f;
        for (int d = 0; d < 10; ++d) {
            float m0 = accum[16 + g0 * 10 + d] / fmaxf(c0, 1.f);
            float v0 = (accum[56 + g0 * 10 + d] - c0 * m0 * m0) / fmaxf(c0 - 1.f, 1.f);
            float m1 = accum[16 + g1 * 10 + d] / fmaxf(c1v, 1.f);
            float v1 = (accum[56 + g1 * 10 + d] - c1v * m1 * m1) / fmaxf(c1v - 1.f, 1.f);
            dsum += (m0 - m1) * (m0 - m1) + (v0 - v1) * (v0 - v1);
        }
        if (c0 >= 2.f && c1v >= 2.f) twin += dsum;
    }
    out[2] = 0.1f * twin;
    float dzx = accum[4] / Nf, dzs = accum[5] / Nf;
    float dxs1 = accum[6] / Nf, dxs2 = accum[7] / Nf;
    out[3] = dzx + expf(-dzs) + expf(-dxs1) + expf(-dxs2);
    out[4] = -accum[8] / (2.f * Nf * 126.f) - accum[9] / (2.f * Nf * 2.f);
    out[5] = -accum[10] / (2.f * Nf * 128.f);
}

// ---------------------------------------------------------------------------
extern "C" void kernel_launch(void* const* d_in, const int* in_sizes, int n_in,
                              void* d_out, int out_size, void* d_ws, size_t ws_size,
                              hipStream_t stream) {
    const float* x   = (const float*)d_in[0];
    const float* a   = (const float*)d_in[1];
    const float* eps = (const float*)d_in[2];
    const int*   y   = (const int*)d_in[3];
    const float* W1  = (const float*)d_in[4];
    const float* b1  = (const float*)d_in[5];
    const float* g1  = (const float*)d_in[6];
    const float* be1 = (const float*)d_in[7];
    const float* g2  = (const float*)d_in[8];
    const float* be2 = (const float*)d_in[9];
    const float* W2  = (const float*)d_in[10];
    const float* b2  = (const float*)d_in[11];
    const float* Wc1 = (const float*)d_in[12];
    const float* bc1 = (const float*)d_in[13];
    const float* Wc2 = (const float*)d_in[14];
    const float* bc2 = (const float*)d_in[15];
    const float* Wd1 = (const float*)d_in[16];
    const float* bd1 = (const float*)d_in[17];
    const float* Wd2 = (const float*)d_in[18];
    const float* bd2 = (const float*)d_in[19];
    const float* Wn  = (const float*)d_in[20];
    const float* bnb = (const float*)d_in[21];

    char* W = (char*)d_ws;
    float*    t1   = (float*)W;                                 // N*128 f32
    _Float16* X1h  = (_Float16*)(W + (size_t)NS * 128 * 4);
    _Float16* X2h  = X1h + (size_t)NS * 128;
    _Float16* Xn1h = X2h + (size_t)NS * 128;
    _Float16* Xn2h = Xn1h + (size_t)NS * 160;
    _Float16* H1h  = Xn2h + (size_t)NS * 160;
    _Float16* H2h  = H1h + (size_t)NS * 128;
    float* E1 = (float*)(H2h + (size_t)NS * 128);
    float* E2 = E1 + (size_t)NS * 48;
    float* E3 = E2 + (size_t)NS * 48;
    float* E4 = E3 + (size_t)NS * 48;
    float* yp = E4 + (size_t)NS * 48;
    float* smallF = yp + NS;          // 4096 stats (4 sets x 1024) + 128 accum
    float* accum  = smallF + 4096;
    _Float16* W1t = (_Float16*)(smallF + 4224);
    _Float16* W2t = W1t + 128 * 128;
    _Float16* Wnt = W2t + 48 * 128;
    float* b2p = (float*)(Wnt + 128 * 160);
    float* WcE = b2p + 48;
    float* bcE = WcE + 20;
    float* WdE = bcE + 2;
    float* bdE = WdE + 2560;

    zero_f32<<<17, 256, 0, stream>>>(smallF, 4224);
    prep_weights<<<1, 256, 0, stream>>>(W1, W2, Wn, b2, Wc1, bc1, Wc2, bc2,
                                        Wd1, bd1, Wd2, bd2,
                                        W1t, W2t, Wnt, b2p, WcE, bcE, WdE, bdE);
    build_X12<<<4096, 256, 0, stream>>>(x, a, X1h, X2h);

    auto enc = [&](const _Float16* Xh, float* Eout, int set) {
        float* st = smallF + set * 1024;
        // st: [0]csum1 [128]csq1 [256]csum2 [384]csq2 [512]a1 [640]c1 [768]a2 [896]c2
        gemm128_stats<<<1024, 256, 0, stream>>>(Xh, W1t, b1, t1, st, st + 128);
        bn_finalize<<<1, 128, 0, stream>>>(st, st + 128, g1, be1, st + 512, st + 640);
        relu_bn_stats<<<2048, 256, 0, stream>>>(t1, st + 512, st + 640, st + 256, st + 384);
        bn_finalize<<<1, 128, 0, stream>>>(st + 256, st + 384, g2, be2, st + 768, st + 896);
        gemm_bn_out<<<1024, 256, 0, stream>>>(t1, st + 512, st + 640, st + 768, st + 896,
                                              W2t, b2p, Eout);
    };

    enc(X1h, E1, 0);
    enc(X2h, E2, 1);
    ypred_kernel<<<NS / 256, 256, 0, stream>>>(E1, E2, eps, WcE, bcE, yp);
    build_Xn<<<4096, 256, 0, stream>>>(x, a, yp, Xn1h, Xn2h);
    gemm160_relu_h<<<1024, 256, 0, stream>>>(Xn1h, Wnt, bnb, H1h);
    gemm160_relu_h<<<1024, 256, 0, stream>>>(Xn2h, Wnt, bnb, H2h);
    enc(H1h, E3, 2);
    enc(H2h, E4, 3);
    final_losses<<<NS / 256, 256, 0, stream>>>(E3, E4, eps, x, a, y,
                                               WdE, bdE, WcE, bcE, accum);
    finalize_out<<<1, 32, 0, stream>>>(accum, (float*)d_out);
}
